// ImitationHybridModel_34986803593283
// MI455X (gfx1250) — matrix operand order repaired and checked by reading, not compiled, not run on hardware
//
#include <hip/hip_runtime.h>

// ---------------------------------------------------------------------------
// ImitationHybridModel forward for MI455X (gfx1250, wave32, WMMA bf16).
// GEMM: 128x128 block tile, 8 waves (2x4), 8 WMMA/wave/k-step, LDS double
// buffered.  A tile: bf16 row-major (padded, b128 loads).  B tile: bf16
// column-major at u32 k-pair granularity so fragments are contiguous b64
// LDS loads.  f32->bf16 conversion happens once at staging (hardware
// v_cvt_pk_bf16_f32 when available), overlapped with compute.
// ---------------------------------------------------------------------------

#define DEV __device__ __forceinline__

typedef __attribute__((ext_vector_type(16))) __bf16 v16bf;
typedef __attribute__((ext_vector_type(8)))  float  v8f;

union FragB16 {
  v16bf v;
  unsigned short u[16];
  unsigned int   w[8];
};

constexpr int kB    = 256;   // batch
constexpr int kD    = 1024;  // model dim
constexpr int kA    = 100;   // slots
constexpr int kH    = 8;     // heads
constexpr int kHD   = 128;   // head dim
constexpr int kAP   = 112;   // slots padded to 16
constexpr int kCols = 96;    // packed head columns (89 real, padded)
#define NEGV (-1e9f)

// ---- fp32 -> bf16 (RNE) ----------------------------------------------------
DEV unsigned short f2bf(float f) {
  unsigned int u = __float_as_uint(f);
  return (unsigned short)((u + 0x7FFFu + ((u >> 16) & 1u)) >> 16);
}
#if __has_builtin(__builtin_amdgcn_cvt_pk_bf16_f32)
DEV unsigned int pk2bf(float lo, float hi) {   // {hi:bf16, lo:bf16} in one op
  auto r = __builtin_amdgcn_cvt_pk_bf16_f32(lo, hi);
  unsigned int u;
  __builtin_memcpy(&u, &r, sizeof(u));
  return u;
}
#else
DEV unsigned int pk2bf(float lo, float hi) {   // {hi:bf16, lo:bf16}
  return (unsigned int)f2bf(lo) | ((unsigned int)f2bf(hi) << 16);
}
#endif

// ---------------------------------------------------------------------------
// Kernel 0: seq-len head (Linear->ReLU->Linear->Sigmoid) + boundary flags.
// ---------------------------------------------------------------------------
__global__ void k_seq_boundary(const float* __restrict__ context,
                               const int*   __restrict__ valid,
                               const float* __restrict__ w1, const float* __restrict__ b1,
                               const float* __restrict__ w2, const float* __restrict__ b2,
                               float* __restrict__ boundary,
                               int*   __restrict__ seq_out) {
  __shared__ float ctxs[kD];
  __shared__ float red[256];
  __shared__ int   lastSh;
  int b = blockIdx.x, t = threadIdx.x;
  for (int d = t; d < kD; d += 256) ctxs[d] = context[b * kD + d];
  __syncthreads();
  float acc = 0.f;
  for (int jj = 0; jj < 2; ++jj) {
    int j = t + jj * 256;                    // coalesced over lanes
    float h = b1[j];
    for (int d = 0; d < kD; ++d) h = fmaf(ctxs[d], w1[d * (kD / 2) + j], h);
    h = fmaxf(h, 0.f);
    acc = fmaf(h, w2[j], acc);
  }
  red[t] = acc;
  __syncthreads();
  for (int s = 128; s > 0; s >>= 1) {
    if (t < s) red[t] += red[t + s];
    __syncthreads();
  }
  if (t == 0) {
    float x   = red[0] + b2[0];
    float sig = 1.f / (1.f + __expf(-x));
    seq_out[b] = (int)rintf(sig * (float)kA);
    int last = -1;
    for (int a = kA - 1; a >= 0; --a) {
      if (valid[b * kA + a] != 0) { last = a; break; }
    }
    lastSh = last;
  }
  __syncthreads();
  for (int a = t; a < kA; a += 256)
    boundary[b * kA + a] = (a == lastSh) ? 1.f : 0.f;
}

// ---------------------------------------------------------------------------
// Kernel 1: comb[b,a,d] = context[b,d] + pos_pad[a,d] + boundary[b,a] -> bf16
// ---------------------------------------------------------------------------
__global__ void k_comb(const float* __restrict__ context,
                       const float* __restrict__ pos,
                       const float* __restrict__ boundary,
                       unsigned short* __restrict__ comb) {
  long long idx = (long long)blockIdx.x * 256 + threadIdx.x;  // < kB*kA*kD
  int d  = (int)(idx % kD);
  int ba = (int)(idx / kD);
  int a  = ba % kA, b = ba / kA;
  float v = context[b * kD + d] + boundary[ba];
  if (d < kD / 4) v += pos[a * (kD / 4) + d];
  comb[idx] = f2bf(v);
}

// ---------------------------------------------------------------------------
// WMMA GEMM:  C[g] (MxN) = A_bf16[g] (MxK=1024) * B_f32[g] (KxN) + bias[g]
// M must be a multiple of 128.
// ---------------------------------------------------------------------------
constexpr int ALD = 20;    // A LDS row stride in u32 (16 data + 4 pad)
constexpr int BCS = 18;    // B LDS column stride in u32 (16 k-pairs + 2 pad)

__global__ __launch_bounds__(256) void k_gemm(
    const unsigned short* __restrict__ Ab, long long lda, long long aGS,
    const float* __restrict__ Bw, int ldb, long long bGS,
    const float* __restrict__ bias, long long biasGS,
    void* __restrict__ Cb, long long ldc, long long cGS,
    int N, int outBf16) {
  __shared__ unsigned int As[2][128 * ALD];   // 2 x 10,240 B (bf16 A tile)
  __shared__ unsigned int Bs[2][128 * BCS];   // 2 x  9,216 B (bf16 B tile, col-major pairs)

  int g = blockIdx.z;
  long long mbase = (long long)blockIdx.y * 128;
  int nbase = blockIdx.x * 128;
  int tid  = threadIdx.x;
  int wave = tid >> 5, lane = tid & 31;
  int mw = (wave >> 2) * 64;                  // wave M offset: 0 / 64
  int nw = (wave & 3) * 32;                   // wave N offset: 0/32/64/96

  const unsigned short* A0 = Ab + g * aGS + mbase * lda;
  const float*          B0 = Bw + g * bGS;
  const bool fullN = (nbase + 128 <= N);      // uniform across the block

  // per-thread staging registers
  uint4 ra0, ra1;
  uint2 rbl[4], rbh[4];                       // B rows 2p (lo) and 2p+1 (hi)

  const int arow_l = tid >> 1;                // A row loaded by this thread
  const int ahalf  = (tid & 1) << 4;          // bf16 element offset 0/16

  auto ldA = [&](int kb) {
    const uint4* s = (const uint4*)(A0 + (long long)arow_l * lda + kb + ahalf);
    ra0 = s[0]; ra1 = s[1];
  };
  auto stA = [&](int buf) {
    uint4* d = (uint4*)&As[buf][arow_l * ALD] + (ahalf >> 3);
    d[0] = ra0; d[1] = ra1;
  };
  // B: each thread loads two b64 from adjacent k rows at one column pair.
  auto ldB = [&](int kb) {
#pragma unroll
    for (int i = 0; i < 4; ++i) {
      int idx = tid + i * 256;                // 0..1023
      int p = idx >> 6;                       // k-pair 0..15
      int c = (idx & 63) << 1;                // column 0..126 (even)
      long long r0 = (long long)(kb + 2 * p) * ldb + nbase + c;
      if (fullN || (nbase + c < N)) {
        rbl[i] = *(const uint2*)(B0 + r0);
        rbh[i] = *(const uint2*)(B0 + r0 + ldb);
      } else {
        rbl[i].x = 0; rbl[i].y = 0; rbh[i].x = 0; rbh[i].y = 0;
      }
    }
  };
  // convert + pack k-pairs; column-major u32 layout -> b64 fragment reads
  auto stB = [&](int buf) {
#pragma unroll
    for (int i = 0; i < 4; ++i) {
      int idx = tid + i * 256;
      int p = idx >> 6;
      int c = (idx & 63) << 1;
      Bs[buf][c * BCS + p] =
          pk2bf(__uint_as_float(rbl[i].x), __uint_as_float(rbh[i].x));
      Bs[buf][(c + 1) * BCS + p] =
          pk2bf(__uint_as_float(rbl[i].y), __uint_as_float(rbh[i].y));
    }
  };

  const int arow  = lane & 15;
  const int khalf = (lane >> 4) << 3;
  const int pB    = (lane >> 4) << 3;         // B k-pair base: 0 / 8
  const int col16 = lane & 15;

  v8f acc[4][2] = {};

  auto compute = [&](int buf) {
    const unsigned int* Au = As[buf];
    const unsigned int* Bu = Bs[buf];
    FragB16 fbv[2];
#pragma unroll
    for (int n = 0; n < 2; ++n) {
      int base = (nw + n * 16 + col16) * BCS + pB;
#pragma unroll
      for (int j = 0; j < 8; ++j)            // 8 consecutive u32 -> b64 loads
        fbv[n].w[j] = Bu[base + j];
    }
#pragma unroll
    for (int m = 0; m < 4; ++m) {
      FragB16 fa;
      int r = mw + m * 16 + arow;
#pragma unroll
      for (int j = 0; j < 8; ++j) {
        int kk = (khalf << 1) + (j << 1);   // contiguous k order (matches B)
        fa.w[j] = Au[r * ALD + (kk >> 1)];
      }
#pragma unroll
      for (int n = 0; n < 2; ++n)
        acc[m][n] = __builtin_amdgcn_wmma_f32_16x16x32_bf16(false, fa.v, false, fbv[n].v,
                                                            (short)0, acc[m][n], false, false);
    }
  };

  // -------- double-buffered main loop (K = 1024, 32 steps) -----------------
  ldA(0); ldB(0);
  stA(0); stB(0);
  __syncthreads();
  for (int s = 0; s < kD / 32; ++s) {
    int cur = s & 1;
    int kn  = (s + 1) * 32;
    if (kn < kD) {
      ldA(kn); ldB(kn);
      if (kn + 32 < kD) {   // deep prefetch into L2 (global_prefetch_b8)
        __builtin_prefetch(A0 + (long long)arow_l * lda + kn + 32 + ahalf, 0, 1);
        __builtin_prefetch(B0 + (long long)(kn + 32 + (tid >> 3)) * ldb + nbase +
                               ((tid & 7) << 4), 0, 1);
      }
    }
    compute(cur);
    if (kn < kD) { stA(cur ^ 1); stB(cur ^ 1); }
    __syncthreads();
  }

  // -------- epilogue -------------------------------------------------------
  int rh = (lane >> 4) << 3;
#pragma unroll
  for (int n = 0; n < 2; ++n) {
    int col = nbase + nw + n * 16 + col16;
    if (col >= N) continue;
    float bvv = bias[g * biasGS + col];
#pragma unroll
    for (int m = 0; m < 4; ++m) {
      long long rbase = mbase + mw + m * 16 + rh;
      if (outBf16) {
        unsigned short* C = (unsigned short*)Cb + g * cGS;
#pragma unroll
        for (int j = 0; j < 8; ++j)
          C[(rbase + j) * ldc + col] = f2bf(acc[m][n][j] + bvv);
      } else {
        float* C = (float*)Cb + g * cGS;
#pragma unroll
        for (int j = 0; j < 8; ++j)
          C[(rbase + j) * ldc + col] = acc[m][n][j] + bvv;
      }
    }
  }
}

// ---------------------------------------------------------------------------
// Attention: one block per (b, h).  QK^T -> mask -> softmax -> PV, all WMMA.
// ---------------------------------------------------------------------------
__global__ void k_attn(const unsigned short* __restrict__ q,
                       const unsigned short* __restrict__ k,
                       const unsigned short* __restrict__ v,
                       const int* __restrict__ valid,
                       unsigned short* __restrict__ attctx) {
  extern __shared__ char smem[];
  unsigned short* Qs = (unsigned short*)smem;          // [112][128] bf16
  unsigned short* Ks = Qs + kAP * kHD;                 // [112][128] bf16
  unsigned short* Vs = Ks + kAP * kHD;                 // [128][128] bf16
  float*          Ss = (float*)(Vs + 128 * kHD);       // [112][128] f32 scores
  unsigned short* Ps = (unsigned short*)(Ss + kAP * kHD); // [112][128] bf16 probs
  int*            Vl = (int*)(Ps + kAP * kHD);         // [112] valid flags

  int bh = blockIdx.x;
  int b = bh / kH, h = bh % kH;
  int tid = threadIdx.x, wave = tid >> 5, lane = tid & 31;

  for (int i = tid; i < kAP * kHD / 2; i += 256) {
    int a = (i * 2) / kHD, d2 = (i * 2) % kHD;
    unsigned int qv = 0, kv = 0;
    if (a < kA) {
      long long base = ((long long)(b * kA + a)) * kD + h * kHD;
      qv = ((const unsigned int*)(q + base))[d2 >> 1];
      kv = ((const unsigned int*)(k + base))[d2 >> 1];
    }
    ((unsigned int*)Qs)[i] = qv;
    ((unsigned int*)Ks)[i] = kv;
  }
  for (int i = tid; i < 128 * kHD / 2; i += 256) {
    int a = (i * 2) / kHD, d2 = (i * 2) % kHD;
    unsigned int vv = 0;
    if (a < kA) {
      long long base = ((long long)(b * kA + a)) * kD + h * kHD;
      vv = ((const unsigned int*)(v + base))[d2 >> 1];
    }
    ((unsigned int*)Vs)[i] = vv;
  }
  for (int i = tid; i < kAP; i += 256) Vl[i] = (i < kA) ? (valid[b * kA + i] != 0) : 0;
  __syncthreads();

  const unsigned int* Qw = (const unsigned int*)Qs;
  const unsigned int* Kw = (const unsigned int*)Ks;
  int arow  = lane & 15;
  int khalf = (lane >> 4) << 3;
  int kbB   = (lane >> 4) << 4;
  int col16 = lane & 15;
  int rh    = (lane >> 4) << 3;

  // ---- scores = Q K^T / sqrt(HD), masked ---------------------------------
  for (int t = wave; t < 49; t += 8) {
    int mt = t / 7, nt = t % 7;
    v8f acc = {};
    for (int kb = 0; kb < kHD; kb += 32) {
      FragB16 fa, fb;
#pragma unroll
      for (int j = 0; j < 8; ++j) {
        int kk = (khalf << 1) + (j << 1);
        fa.w[j] = Qw[((mt * 16 + arow) * kHD + kb + kk) >> 1];
      }
#pragma unroll
      for (int j = 0; j < 8; ++j)
        fb.w[j] = Kw[((nt * 16 + col16) * kHD + kb + kbB + 2 * j) >> 1];
      acc = __builtin_amdgcn_wmma_f32_16x16x32_bf16(false, fa.v, false, fb.v,
                                                    (short)0, acc, false, false);
    }
    int ca = nt * 16 + col16;
#pragma unroll
    for (int j = 0; j < 8; ++j) {
      int ra = mt * 16 + rh + j;
      float s;
      if (ra >= kA || ca >= kA)      s = -INFINITY;       // padding
      else if (Vl[ra] && Vl[ca])     s = NEGV;            // faithful mask bug
      else                           s = acc[j] * 0.08838834764831845f;
      Ss[ra * kHD + ca] = s;
    }
  }
  __syncthreads();

  // ---- softmax rows -> bf16 P (zero K-pad cols 112..127) ------------------
  for (int r = wave; r < kAP; r += 8) {
    if (r < kA) {
      float mx = -INFINITY;
      for (int c = lane; c < kAP; c += 32) mx = fmaxf(mx, Ss[r * kHD + c]);
      for (int off = 16; off > 0; off >>= 1) mx = fmaxf(mx, __shfl_xor(mx, off, 32));
      float sum = 0.f;
      for (int c = lane; c < kAP; c += 32) sum += __expf(Ss[r * kHD + c] - mx);
      for (int off = 16; off > 0; off >>= 1) sum += __shfl_xor(sum, off, 32);
      float inv = 1.f / sum;
      for (int c = lane; c < kAP; c += 32)
        Ps[r * kHD + c] = f2bf(__expf(Ss[r * kHD + c] - mx) * inv);
      for (int c = kAP + lane; c < 128; c += 32) Ps[r * kHD + c] = 0;
    } else {
      for (int c = lane; c < 128; c += 32) Ps[r * kHD + c] = 0;
    }
  }
  __syncthreads();

  // ---- ctx = P @ V --------------------------------------------------------
  const unsigned int* Pw = (const unsigned int*)Ps;
  for (int t = wave; t < 56; t += 8) {
    int mt = t / 8, dt = t % 8;
    v8f acc = {};
    for (int kb = 0; kb < 128; kb += 32) {
      FragB16 fa, fb;
#pragma unroll
      for (int j = 0; j < 8; ++j) {
        int kk = (khalf << 1) + (j << 1);
        fa.w[j] = Pw[((mt * 16 + arow) * kHD + kb + kk) >> 1];
      }
#pragma unroll
      for (int j = 0; j < 8; ++j) {
        int k0 = kb + kbB + 2 * j;
        fb.u[2 * j]     = Vs[k0 * kHD + dt * 16 + col16];
        fb.u[2 * j + 1] = Vs[(k0 + 1) * kHD + dt * 16 + col16];
      }
      acc = __builtin_amdgcn_wmma_f32_16x16x32_bf16(false, fa.v, false, fb.v,
                                                    (short)0, acc, false, false);
    }
#pragma unroll
    for (int j = 0; j < 8; ++j) {
      int ra = mt * 16 + rh + j;
      if (ra < kA)
        attctx[((long long)(b * kA + ra)) * kD + h * kHD + dt * 16 + col16] = f2bf(acc[j]);
    }
  }
}

// ---------------------------------------------------------------------------
// LayerNorm + ReLU over each (b,a) row of h -> feat bf16.  Block per row.
// ---------------------------------------------------------------------------
__global__ void k_ln(const float* __restrict__ hbuf,
                     const float* __restrict__ gam,
                     const float* __restrict__ bet,
                     unsigned short* __restrict__ feat) {
  __shared__ float s1[256], s2[256];
  long long row = blockIdx.x;                   // < kB*kA
  int a = (int)(row % kA);
  int t = threadIdx.x;
  const float* hr = hbuf + row * kD;
  float loc[4];
  float sum = 0.f, sq = 0.f;
#pragma unroll
  for (int i = 0; i < 4; ++i) {
    float x = hr[t + i * 256];
    loc[i] = x; sum += x; sq += x * x;
  }
  s1[t] = sum; s2[t] = sq;
  __syncthreads();
  for (int s = 128; s > 0; s >>= 1) {
    if (t < s) { s1[t] += s1[t + s]; s2[t] += s2[t + s]; }
    __syncthreads();
  }
  float mean = s1[0] * (1.f / kD);
  float var  = s2[0] * (1.f / kD) - mean * mean;
  float rstd = rsqrtf(var + 1e-5f);
#pragma unroll
  for (int i = 0; i < 4; ++i) {
    int d = t + i * 256;
    float y = (loc[i] - mean) * rstd * gam[a * kD + d] + bet[a * kD + d];
    feat[row * kD + d] = f2bf(fmaxf(y, 0.f));
  }
}

// ---------------------------------------------------------------------------
// Pack the 10 small heads into one [kD x kCols] f32 weight + [kCols] bias.
// ---------------------------------------------------------------------------
__global__ void k_packhead(const float* ew, const float* eb, const float* tw, const float* tb,
                           const float* xmw, const float* xmb, const float* xlw, const float* xlb,
                           const float* ymw, const float* ymb, const float* ylw, const float* ylb,
                           const float* bw, const float* bb, const float* kaw, const float* kab,
                           const float* kw, const float* kb2, const float* sw, const float* sb,
                           float* __restrict__ headw, float* __restrict__ headb) {
  int idx = blockIdx.x * 256 + threadIdx.x;
  if (idx >= kD * kCols) return;
  int kk = idx / kCols, c = idx % kCols;
  float wv = 0.f, bv = 0.f;
  if (c < 8)        { wv = ew[kk * 8 + c];        bv = eb[c]; }
  else if (c < 11)  { wv = tw[kk * 3 + (c - 8)];  bv = tb[c - 8]; }
  else if (c == 11) { wv = xmw[kk];               bv = xmb[0]; }
  else if (c == 12) { wv = xlw[kk];               bv = xlb[0]; }
  else if (c == 13) { wv = ymw[kk];               bv = ymb[0]; }
  else if (c == 14) { wv = ylw[kk];               bv = ylb[0]; }
  else if (c < 19)  { wv = bw[kk * 4 + (c - 15)]; bv = bb[c - 15]; }
  else if (c < 22)  { wv = kaw[kk * 3 + (c - 19)];bv = kab[c - 19]; }
  else if (c < 86)  { wv = kw[kk * 64 + (c - 22)];bv = kb2[c - 22]; }
  else if (c < 89)  { wv = sw[kk * 3 + (c - 86)]; bv = sb[c - 86]; }
  headw[idx] = wv;
  if (kk == 0) headb[c] = bv;
}

// ---------------------------------------------------------------------------
// Final epilogue: masks / sigmoid / softplus, scatter into d_out segments.
// ---------------------------------------------------------------------------
__global__ void k_epilogue(const float* __restrict__ ho,
                           const int* __restrict__ valid,
                           float* __restrict__ out) {
  int idx = blockIdx.x * 256 + threadIdx.x;     // < kB*kA
  if (idx >= kB * kA) return;
  bool inval = (valid[idx] == 0);
  const float* r = ho + (long long)idx * kCols;
  float* o_event = out;                 // [B,A,8]
  float* o_time  = out + 204800;        // [B,A,3]
  float* o_xmu   = out + 281600;        // [B,A]
  float* o_xls   = out + 307200;
  float* o_ymu   = out + 332800;
  float* o_yls   = out + 358400;
  float* o_btn   = out + 384000;        // [B,A,4]
  float* o_ka    = out + 486400;        // [B,A,3]
  float* o_kid   = out + 563200;        // [B,A,64]
  float* o_scr   = out + 2201600;       // [B,A,3]
  for (int e = 0; e < 8; ++e) o_event[idx * 8 + e] = inval ? NEGV : r[e];
  for (int e = 0; e < 3; ++e) {
    float x  = r[8 + e];
    float sp = (x > 20.f) ? x : log1pf(__expf(x));
    o_time[idx * 3 + e] = sp + 0.001f;
  }
  o_xmu[idx] = inval ? 0.5f : 1.f / (1.f + __expf(-r[11]));
  o_xls[idx] = inval ? -2.f : r[12];
  o_ymu[idx] = inval ? 0.5f : 1.f / (1.f + __expf(-r[13]));
  o_yls[idx] = inval ? -2.f : r[14];
  for (int e = 0; e < 4; ++e)  o_btn[idx * 4 + e]  = inval ? NEGV : r[15 + e];
  for (int e = 0; e < 3; ++e)  o_ka[idx * 3 + e]   = inval ? NEGV : r[19 + e];
  for (int e = 0; e < 64; ++e) o_kid[idx * 64 + e] = inval ? NEGV : r[22 + e];
  for (int e = 0; e < 3; ++e)  o_scr[idx * 3 + e]  = inval ? NEGV : r[86 + e];
}

// ---------------------------------------------------------------------------
// Workspace layout (aliased; every region fully written before read).
// Requires ws_size >= ~263 MB.
// ---------------------------------------------------------------------------
constexpr size_t SZ_BAD     = (size_t)kB * kA * kD * 2;           // 52,428,800 B
constexpr size_t WS_COMB    = 0;                 // bf16 [B*A, D]
constexpr size_t WS_Q       = 1 * SZ_BAD;        // bf16 [B*A, D]
constexpr size_t WS_K       = 2 * SZ_BAD;        // bf16 [B*A, D]
constexpr size_t WS_V       = 3 * SZ_BAD;        // bf16 [B*A, D]
constexpr size_t WS_ATTCTX  = 4 * SZ_BAD;        // bf16 [B*A, D]
constexpr size_t WS_ATTOUT  = WS_Q;              // reuse (q dead)
constexpr size_t WS_H       = WS_K;              // f32 [B*A, D] spans K+V (dead)
constexpr size_t WS_FEAT    = WS_COMB;           // reuse (comb dead)
constexpr size_t WS_HEADOUT = WS_ATTCTX;         // f32 [B*A, 96] (attctx dead)
constexpr size_t WS_HEADW   = 5 * SZ_BAD;        // f32 [D, 96]
constexpr size_t WS_HEADB   = WS_HEADW + (size_t)kD * kCols * 4;
constexpr size_t WS_BOUND   = WS_HEADB + 512;    // f32 [B*A]

extern "C" void kernel_launch(void* const* d_in, const int* in_sizes, int n_in,
                              void* d_out, int out_size, void* d_ws, size_t ws_size,
                              hipStream_t stream) {
  const float* context = (const float*)d_in[0];
  const int*   valid   = (const int*)  d_in[1];
  const float* sl_w1 = (const float*)d_in[2];  const float* sl_b1 = (const float*)d_in[3];
  const float* sl_w2 = (const float*)d_in[4];  const float* sl_b2 = (const float*)d_in[5];
  const float* pos_emb = (const float*)d_in[6];
  const float* wq = (const float*)d_in[7];   const float* bq = (const float*)d_in[8];
  const float* wk = (const float*)d_in[9];   const float* bk = (const float*)d_in[10];
  const float* wv = (const float*)d_in[11];  const float* bv = (const float*)d_in[12];
  const float* wo = (const float*)d_in[13];  const float* bo = (const float*)d_in[14];
  const float* proc_w = (const float*)d_in[15]; const float* proc_b = (const float*)d_in[16];
  const float* ln_g = (const float*)d_in[17];   const float* ln_b = (const float*)d_in[18];
  const float* event_w = (const float*)d_in[19]; const float* event_b = (const float*)d_in[20];
  const float* timeq_w = (const float*)d_in[21]; const float* timeq_b = (const float*)d_in[22];
  const float* xmu_w = (const float*)d_in[23]; const float* xmu_b = (const float*)d_in[24];
  const float* xls_w = (const float*)d_in[25]; const float* xls_b = (const float*)d_in[26];
  const float* ymu_w = (const float*)d_in[27]; const float* ymu_b = (const float*)d_in[28];
  const float* yls_w = (const float*)d_in[29]; const float* yls_b = (const float*)d_in[30];
  const float* btn_w = (const float*)d_in[31]; const float* btn_b = (const float*)d_in[32];
  const float* ka_w  = (const float*)d_in[33]; const float* ka_b  = (const float*)d_in[34];
  const float* kid_w = (const float*)d_in[35]; const float* kid_b = (const float*)d_in[36];
  const float* scr_w = (const float*)d_in[37]; const float* scr_b = (const float*)d_in[38];
  (void)in_sizes; (void)n_in; (void)out_size; (void)ws_size;

  char* ws = (char*)d_ws;
  unsigned short* comb    = (unsigned short*)(ws + WS_COMB);
  unsigned short* qb      = (unsigned short*)(ws + WS_Q);
  unsigned short* kbuf    = (unsigned short*)(ws + WS_K);
  unsigned short* vbuf    = (unsigned short*)(ws + WS_V);
  unsigned short* attctx  = (unsigned short*)(ws + WS_ATTCTX);
  unsigned short* attout  = (unsigned short*)(ws + WS_ATTOUT);
  float*          hbuf    = (float*)(ws + WS_H);
  unsigned short* feat    = (unsigned short*)(ws + WS_FEAT);
  float*          headout = (float*)(ws + WS_HEADOUT);
  float*          headw   = (float*)(ws + WS_HEADW);
  float*          headb   = (float*)(ws + WS_HEADB);
  float*          boundary= (float*)(ws + WS_BOUND);

  float* out   = (float*)d_out;
  int*   o_seq = (int*)(out + 2278400);

  // 0: seq_len + boundary
  k_seq_boundary<<<kB, 256, 0, stream>>>(context, valid, sl_w1, sl_b1, sl_w2, sl_b2,
                                         boundary, o_seq);
  // 1: comb (bf16)
  k_comb<<<(kB * kA * kD) / 256, 256, 0, stream>>>(context, pos_emb, boundary, comb);

  // 2-4: Q, K, V projections (WMMA): M = 25600, N = 1024
  dim3 gProj(kD / 128, (kB * kA) / 128, 1);
  k_gemm<<<gProj, 256, 0, stream>>>(comb, kD, 0, wq, kD, 0, bq, 0, qb,   kD, 0, kD, 1);
  k_gemm<<<gProj, 256, 0, stream>>>(comb, kD, 0, wk, kD, 0, bk, 0, kbuf, kD, 0, kD, 1);
  k_gemm<<<gProj, 256, 0, stream>>>(comb, kD, 0, wv, kD, 0, bv, 0, vbuf, kD, 0, kD, 1);

  // 5: fused attention per (b, h)
  size_t attShm = (size_t)kAP * kHD * 2 * 2      // Q + K
                + (size_t)128 * kHD * 2          // V
                + (size_t)kAP * kHD * 4          // scores f32
                + (size_t)kAP * kHD * 2          // P bf16
                + (size_t)kAP * 4;               // valid flags
  k_attn<<<kB * kH, 256, attShm, stream>>>(qb, kbuf, vbuf, valid, attctx);

  // 6: output projection (WMMA)
  k_gemm<<<gProj, 256, 0, stream>>>(attctx, kD, 0, wo, kD, 0, bo, 0, attout, kD, 0, kD, 1);

  // 7: pack head weights
  k_packhead<<<(kD * kCols + 255) / 256, 256, 0, stream>>>(
      event_w, event_b, timeq_w, timeq_b, xmu_w, xmu_b, xls_w, xls_b,
      ymu_w, ymu_b, yls_w, yls_b, btn_w, btn_b, ka_w, ka_b,
      kid_w, kid_b, scr_w, scr_b, headw, headb);

  // 8: grouped per-slot GEMM (100 groups, WMMA): M = 256 per group
  dim3 gProc(kD / 128, kB / 128, kA);
  k_gemm<<<gProc, 256, 0, stream>>>(attout, (long long)kA * kD, kD,
                                    proc_w, kD, (long long)kD * kD,
                                    proc_b, kD,
                                    hbuf, (long long)kA * kD, kD,
                                    kD, 0);

  // 9: LayerNorm + ReLU -> feat (bf16)
  k_ln<<<kB * kA, 256, 0, stream>>>(hbuf, ln_g, ln_b, feat);

  // 10: heads GEMM (N = 96 padded): M = 25600
  dim3 gHead(1, (kB * kA) / 128, 1);
  k_gemm<<<gHead, 256, 0, stream>>>(feat, kD, 0, headw, kCols, 0, headb, 0,
                                    headout, kCols, 0, kCols, 0);

  // 11: masked epilogue -> d_out
  k_epilogue<<<(kB * kA + 255) / 256, 256, 0, stream>>>(headout, valid, out);
}
